// DynamicUpsampling_8675833938531
// MI455X (gfx1250) — matrix operand order repaired and checked by reading, not compiled, not run on hardware
//
#include <hip/hip_runtime.h>

typedef __attribute__((ext_vector_type(16))) __bf16 v16bf;
typedef __attribute__((ext_vector_type(8)))  __bf16 v8bf;
typedef __attribute__((ext_vector_type(8)))  float  v8f;

#define DS    4
#define CFEA  32
#define NB    2
#define HLR   128
#define WLR   256
#define HP    (HLR + 2)    // 130 padded rows
#define WP    (WLR + 2)    // 258 padded cols
#define K2    144
#define CIN1  513
#define BN_EPS 1e-5f
#define OCT   (K2 / 16)    // 9 out-channel tiles
#define XT2   (WLR / 32)   // 8 double pixel tiles per row
#define HW    (HLR * WLR)  // 32768
#define KT1   16           // conv1 input k-tiles (512 feature channels)
#define KT23  5            // conv2/3 input k-tiles (144 ch padded to 160)

// ---------------------------------------------------------------------------
// zero fill (uint4 granularity, graph-capture safe)
// ---------------------------------------------------------------------------
__global__ __launch_bounds__(256) void zfill_kernel(uint4* __restrict__ p, size_t n16)
{
    size_t i = (size_t)blockIdx.x * 256 + threadIdx.x;
    if (i < n16) p[i] = make_uint4(0u, 0u, 0u, 0u);
}

// ---------------------------------------------------------------------------
// left_fea [B,32,512,1024] f32 -> x0 interleaved [b][kt=16][HP][WP][32] bf16
// channel cf = c*16 + i*4 + j ; kt = cf>>5, c32 = cf&31. Reads fully coalesced.
// ---------------------------------------------------------------------------
__global__ __launch_bounds__(256) void pack_x0_kernel(
    const float* __restrict__ lf, __bf16* __restrict__ x0)
{
    size_t o = (size_t)blockIdx.x * 256 + threadIdx.x;   // exactly B*32*512*1024
    int xf = (int)(o & 1023); size_t u = o >> 10;
    int yf = (int)(u & 511);  u >>= 9;
    int c  = (int)(u & 31);   int b = (int)(u >> 5);
    int x = xf >> 2, j = xf & 3;
    int y = yf >> 2, i = yf & 3;
    int cf = c * 16 + i * 4 + j;
    int kt = cf >> 5, c32 = cf & 31;
    float v = lf[o];
    x0[((((size_t)(b * KT1 + kt)) * HP + (y + 1)) * WP + (x + 1)) * 32 + c32] = (__bf16)v;
}

// ---------------------------------------------------------------------------
// disp [B,H,W] -> padded [B,HP,WP] f32, zero border
// ---------------------------------------------------------------------------
__global__ __launch_bounds__(256) void pad_disp_kernel(
    const float* __restrict__ disp, float* __restrict__ dp)
{
    int o = blockIdx.x * 256 + threadIdx.x;
    const int N = NB * HP * WP;
    if (o >= N) return;
    int px = o % WP; int u = o / WP;
    int py = u % HP; int b = u / HP;
    float v = 0.f;
    if (py >= 1 && py <= HLR && px >= 1 && px <= WLR)
        v = disp[((size_t)b * HLR + (py - 1)) * WLR + (px - 1)];
    dp[o] = v;
}

// ---------------------------------------------------------------------------
// Weight prepack into WMMA A-fragment layout, fragment order [oct][kt][tap]
// so per-octile A offsets fit in 24-bit instruction immediates.
// A 16x32 bf16: m = lane&15 ; k = h + (lane>=16?16:0)
// ---------------------------------------------------------------------------
template<int KTILES, int CINW, int CHOFF>
__global__ __launch_bounds__(256) void pack_w_kernel(
    const float* __restrict__ w, __bf16* __restrict__ wp)
{
    int p = blockIdx.x * 256 + threadIdx.x;              // exact multiple of 256
    int h    = p & 15;
    int lane = (p >> 4) & 31;
    int frag = p >> 9;
    int tap = frag % 9;
    int kt  = (frag / 9) % KTILES;
    int oct = frag / (9 * KTILES);
    int m  = lane & 15;
    int hi = lane >> 4;
    int k  = h + (hi ? 16 : 0);
    int oc = oct * 16 + m;
    int ch = kt * 32 + k + CHOFF;
    float v = 0.f;
    if (ch < CINW) v = w[((size_t)oc * CINW + ch) * 9 + tap];
    wp[p] = (__bf16)v;
}

// conv1 disparity-column weights -> dense [oct][tap][16] f32
__global__ __launch_bounds__(256) void pack_wd_kernel(
    const float* __restrict__ w1, float* __restrict__ wd)
{
    int p = blockIdx.x * 256 + threadIdx.x;
    if (p >= OCT * 9 * 16) return;
    int m = p & 15; int tap = (p >> 4) % 9; int oct = p / (16 * 9);
    wd[p] = w1[(size_t)(oct * 16 + m) * (CIN1 * 9) + tap];
}

// ---------------------------------------------------------------------------
// Implicit-GEMM conv3x3 + BN (+ReLU) via WMMA bf16.
// One wave = 16 outch x 32 pixels: one A fragment shared by two B fragments,
// two independent accumulator chains. kt-outer loop; all tap offsets are
// instruction immediates. All loads unconditional (halo/pad in buffers).
// ---------------------------------------------------------------------------
template<int KTIN, bool HAS_DISP, bool RELU, bool OUT_BF16, int KTOUT>
__global__ __launch_bounds__(256) void conv_stage_kernel(
    const __bf16* __restrict__ xin,   // [b][KTIN][HP][WP][32] bf16
    const float*  __restrict__ dp,    // [b][HP][WP] f32 padded disp
    const __bf16* __restrict__ wpack, // packed A fragments [oct][kt][tap]
    const float*  __restrict__ wd,    // [oct][tap][16] disp weights
    const float*  __restrict__ g, const float* __restrict__ bias,
    __bf16* __restrict__ out_bf,      // [b][KTOUT][HP][WP][32] bf16
    float*  __restrict__ out_f)       // [b][K2][H][W] f32
{
    int lane = threadIdx.x & 31;
    int wav  = threadIdx.x >> 5;
    int t = blockIdx.x * 8 + wav;                        // 18432 tiles, exact
    int octile = t % OCT; int u = t / OCT;
    int xt2 = u % XT2; u /= XT2;
    int y  = u % HLR;
    int b  = u / HLR;
    int px0 = xt2 * 32;
    int n  = lane & 15;
    int hi = lane >> 4;

    const __bf16* xbase = xin
        + (((size_t)b * KTIN * HP + y) * WP + (px0 + n)) * 32 + (hi << 4);
    const __bf16* wbase = wpack + (size_t)octile * (KTIN * 9 * 512) + lane * 16;
    const float*  dbase = HAS_DISP ? (dp + ((size_t)b * HP + y) * WP + px0 + n) : nullptr;

    v8f acc0 = {}, acc1 = {};
    #pragma unroll
    for (int kt = 0; kt < KTIN; ++kt) {
        const __bf16* xk = xbase + (size_t)kt * (HP * WP * 32);
        #pragma unroll
        for (int tap = 0; tap < 9; ++tap) {
            const int dy = tap / 3, dx = tap % 3;
            v16bf a  = *(const v16bf*)(wbase + (kt * 9 + tap) * 512);
            v16bf b0 = *(const v16bf*)(xk + (dy * WP + dx) * 32);
            v16bf b1 = *(const v16bf*)(xk + (dy * WP + dx) * 32 + 512);  // +16 px
            acc0 = __builtin_amdgcn_wmma_f32_16x16x32_bf16(
                false, a, false, b0, (short)0, acc0, false, false);
            acc1 = __builtin_amdgcn_wmma_f32_16x16x32_bf16(
                false, a, false, b1, (short)0, acc1, false, false);
        }
    }

    if (HAS_DISP) {  // rank-1 update for input channel 0 (disparity)
        #pragma unroll
        for (int tap = 0; tap < 9; ++tap) {
            const int dy = tap / 3, dx = tap % 3;
            float dv0 = dbase[dy * WP + dx];
            float dv1 = dbase[dy * WP + dx + 16];
            #pragma unroll
            for (int v = 0; v < 8; ++v) {
                float wv = wd[(octile * 9 + tap) * 16 + v + (hi << 3)];
                acc0[v] += wv * dv0;
                acc1[v] += wv * dv1;
            }
        }
    }

    const float bninv = rsqrtf(1.f + BN_EPS);
    int oc0 = octile * 16 + (hi << 3);                   // 8 consecutive channels
    float sc[8], bb[8];
    #pragma unroll
    for (int v = 0; v < 8; ++v) { sc[v] = g[oc0 + v] * bninv; bb[v] = bias[oc0 + v]; }

    if (OUT_BF16) {
        v8bf ov0, ov1;
        #pragma unroll
        for (int v = 0; v < 8; ++v) {
            float v0 = acc0[v] * sc[v] + bb[v];
            float v1 = acc1[v] * sc[v] + bb[v];
            if (RELU) { v0 = fmaxf(v0, 0.f); v1 = fmaxf(v1, 0.f); }
            ov0[v] = (__bf16)v0; ov1[v] = (__bf16)v1;
        }
        __bf16* optr = out_bf
            + (((size_t)(b * KTOUT + (oc0 >> 5)) * HP + (y + 1)) * WP + (px0 + n + 1)) * 32
            + (oc0 & 31);
        *(v8bf*)optr = ov0;                              // one 16B store per lane
        *(v8bf*)(optr + 16 * 32) = ov1;                  // +16 px
    } else {
        #pragma unroll
        for (int v = 0; v < 8; ++v) {
            float v0 = acc0[v] * sc[v] + bb[v];
            float v1 = acc1[v] * sc[v] + bb[v];
            if (RELU) { v0 = fmaxf(v0, 0.f); v1 = fmaxf(v1, 0.f); }
            size_t oidx = ((size_t)b * K2 + oc0 + v) * HW + (size_t)y * WLR + px0 + n;
            out_f[oidx]      = v0;
            out_f[oidx + 16] = v1;
        }
    }
}

// ---------------------------------------------------------------------------
// Softmax over 9 taps x 16 groups, blend replication-padded disparity,
// pixel-shuffle 4x, scale by DS.
// ---------------------------------------------------------------------------
__global__ __launch_bounds__(256) void fuse_out_kernel(
    const float* __restrict__ x3, const float* __restrict__ disp,
    float* __restrict__ out)
{
    int gid = blockIdx.x * 256 + threadIdx.x;            // B*HW, exact
    int x = gid % WLR; int u = gid / WLR;
    int y = u % HLR;
    int b = u / HLR;

    const float* db = disp + (size_t)b * HW;
    float content[9];
    #pragma unroll
    for (int di = 0; di < 3; ++di)
        #pragma unroll
        for (int dj = 0; dj < 3; ++dj) {
            int py = min(max(y + di - 1, 0), HLR - 1);
            int px = min(max(x + dj - 1, 0), WLR - 1);
            content[di * 3 + dj] = db[(size_t)py * WLR + px];
        }

    const float* xb = x3 + (size_t)b * K2 * HW + (size_t)y * WLR + x;
    #pragma unroll
    for (int gidx = 0; gidx < 16; ++gidx) {
        float L[9], mx = -1e30f;
        #pragma unroll
        for (int t = 0; t < 9; ++t) {
            L[t] = xb[(size_t)(gidx * 9 + t) * HW];
            mx = fmaxf(mx, L[t]);
        }
        float s = 0.f, r = 0.f;
        #pragma unroll
        for (int t = 0; t < 9; ++t) {
            float e = __expf(L[t] - mx);
            s += e;
            r += e * content[t];
        }
        float res = (r / s) * (float)DS;
        int gi = gidx >> 2, gj = gidx & 3;
        out[((size_t)b * (HLR * DS) + (y * DS + gi)) * (size_t)(WLR * DS) + (x * DS + gj)] = res;
    }
}

// ---------------------------------------------------------------------------
extern "C" void kernel_launch(void* const* d_in, const int* in_sizes, int n_in,
                              void* d_out, int out_size, void* d_ws, size_t ws_size,
                              hipStream_t stream) {
    const float* disp = (const float*)d_in[0];
    const float* lf   = (const float*)d_in[1];
    const float* w1   = (const float*)d_in[2];
    const float* g1   = (const float*)d_in[3];
    const float* b1   = (const float*)d_in[4];
    const float* w2   = (const float*)d_in[5];
    const float* g2   = (const float*)d_in[6];
    const float* b2   = (const float*)d_in[7];
    const float* w3   = (const float*)d_in[8];
    const float* g3   = (const float*)d_in[9];
    const float* b3   = (const float*)d_in[10];
    float* out = (float*)d_out;

    size_t off = 0;
    auto carve = [&](size_t bytes) { size_t o = off; off = (off + bytes + 255) & ~(size_t)255; return o; };
    const size_t n_x0 = (size_t)NB * KT1  * HP * WP * 32;   // bf16
    const size_t n_x12 = (size_t)NB * KT23 * HP * WP * 32;  // bf16
    const size_t n_x3 = (size_t)NB * K2 * HW;               // f32
    const size_t n_wp1 = (size_t)OCT * KT1  * 9 * 512;      // bf16
    const size_t n_wp23 = (size_t)OCT * KT23 * 9 * 512;     // bf16
    const size_t n_dp = (size_t)NB * HP * WP;               // f32
    char* ws = (char*)d_ws;
    __bf16* x0  = (__bf16*)(ws + carve(n_x0  * 2));
    __bf16* x1  = (__bf16*)(ws + carve(n_x12 * 2));
    __bf16* x2  = (__bf16*)(ws + carve(n_x12 * 2));
    float*  x3  = (float*) (ws + carve(n_x3  * 4));
    __bf16* wp1 = (__bf16*)(ws + carve(n_wp1 * 2));
    __bf16* wp2 = (__bf16*)(ws + carve(n_wp23 * 2));
    __bf16* wp3 = (__bf16*)(ws + carve(n_wp23 * 2));
    float*  wd1 = (float*) (ws + carve((size_t)OCT * 9 * 16 * 4));
    float*  dp  = (float*) (ws + carve(n_dp * 4));

    // 0) zero-fill interleaved activation buffers (halo + channel padding)
    auto zf = [&](void* p, size_t bytes) {
        size_t n16 = bytes / 16;
        zfill_kernel<<<(unsigned)((n16 + 255) / 256), 256, 0, stream>>>((uint4*)p, n16);
    };
    zf(x0, n_x0 * 2);
    zf(x1, n_x12 * 2);
    zf(x2, n_x12 * 2);

    // 1) input packing + weight prepack
    pack_x0_kernel<<<(unsigned)(((size_t)NB * CFEA * 512 * 1024) / 256), 256, 0, stream>>>(lf, x0);
    pad_disp_kernel<<<(unsigned)(((size_t)NB * HP * WP + 255) / 256), 256, 0, stream>>>(disp, dp);
    pack_w_kernel<KT1,  CIN1, 1><<<(unsigned)(n_wp1  / 256), 256, 0, stream>>>(w1, wp1);
    pack_w_kernel<KT23, K2,   0><<<(unsigned)(n_wp23 / 256), 256, 0, stream>>>(w2, wp2);
    pack_w_kernel<KT23, K2,   0><<<(unsigned)(n_wp23 / 256), 256, 0, stream>>>(w3, wp3);
    pack_wd_kernel<<<(unsigned)((OCT * 9 * 16 + 255) / 256), 256, 0, stream>>>(w1, wd1);

    // 2) conv stack: 18432 tiles (16oc x 32px), 8 waves/block -> 2304 blocks
    const unsigned cblocks = (unsigned)((size_t)NB * HLR * XT2 * OCT / 8);
    conv_stage_kernel<KT1,  true,  true,  true,  KT23><<<cblocks, 256, 0, stream>>>(
        x0, dp, wp1, wd1, g1, b1, x1, nullptr);
    conv_stage_kernel<KT23, false, true,  true,  KT23><<<cblocks, 256, 0, stream>>>(
        x1, nullptr, wp2, nullptr, g2, b2, x2, nullptr);
    conv_stage_kernel<KT23, false, false, false, 1><<<cblocks, 256, 0, stream>>>(
        x2, nullptr, wp3, nullptr, g3, b3, nullptr, x3);

    // 3) softmax blend + pixel shuffle
    fuse_out_kernel<<<(unsigned)((size_t)NB * HW / 256), 256, 0, stream>>>(x3, disp, out);
}
